// I2ST_49374944035211
// MI455X (gfx1250) — matrix operand order repaired and checked by reading, not compiled, not run on hardware
//
#include <hip/hip_runtime.h>
#include <hip/hip_bf16.h>

typedef __attribute__((ext_vector_type(16))) _Float16 v16h;
typedef __attribute__((ext_vector_type(8)))  _Float16 v8h;
typedef __attribute__((ext_vector_type(8)))  float    v8f;

#define NQTOT 32768
#define CDIM  128
#define NP    128     // padded instance count
#define NREAL 100
#define IMH   48
#define IMW   160
#define HWPIX 7680
#define NCLS  21
#define FFND  512

enum { GF_BIAS = 1, GF_GELU = 2, GF_RES = 4, GF_RELU = 8, GF_TRANSC = 16 };

// ---------------------------------------------------------------------------
// Generic WMMA GEMM:  C[M,N] = alpha * A[M,K] @ B[N,K]^T  (+bias[n]) (+res)
// One wave computes a 16 x (16*NB) slab.  Per K-step: load A fragment once,
// load ALL NB B fragments (distinct registers, clause-able), then issue NB
// independent V_WMMA_F32_16X16X32_F16 (partial loadcnt waits, no D->A/B
// hazards).  A/B fragments use contiguous k = 16*(lane/16) + s (wave32).
// ---------------------------------------------------------------------------
template <int NB>
__global__ __launch_bounds__(256)
void k_gemm_wmma(const _Float16* __restrict__ A, int lda,
                 const _Float16* __restrict__ B, int ldb,
                 const float* __restrict__ bias,
                 const float* __restrict__ res, int ldres,
                 float* __restrict__ C, int ldc,
                 _Float16* __restrict__ Ch, int ldch,
                 int M, int K, int tilesNG, int Nvalid,
                 float alpha, int flags)
{
    int lane = threadIdx.x & 31;
    int wid  = threadIdx.x >> 5;
    int tilesM = M >> 4;
    int tile = blockIdx.x * (blockDim.x >> 5) + wid;
    if (tile >= tilesM * tilesNG) return;          // wave-uniform exit
    int mt = tile / tilesNG, ng = tile % tilesNG;
    int lr = lane & 15, hi = lane >> 4;

    const _Float16* arow = A + (size_t)(mt * 16 + lr) * lda + 16 * hi;
    const _Float16* brow[NB];
#pragma unroll
    for (int j = 0; j < NB; ++j)
        brow[j] = B + (size_t)((ng * NB + j) * 16 + lr) * ldb + 16 * hi;

    v8f acc[NB];
#pragma unroll
    for (int j = 0; j < NB; ++j) acc[j] = (v8f){};

    for (int kb = 0; kb < K; kb += 32) {
        __builtin_prefetch(arow + kb + 128, 0, 3);
        v8h a0 = *(const v8h*)(arow + kb);
        v8h a1 = *(const v8h*)(arow + kb + 8);
        v16h bv[NB];
#pragma unroll
        for (int j = 0; j < NB; ++j) bv[j] = *(const v16h*)(brow[j] + kb);
        v16h av = {};
#pragma unroll
        for (int i = 0; i < 8; ++i) { av[i] = a0[i]; av[8 + i] = a1[i]; }
#pragma unroll
        for (int j = 0; j < NB; ++j)
            acc[j] = __builtin_amdgcn_wmma_f32_16x16x32_f16(false, av, false, bv[j],
                                                            (short)0, acc[j],
                                                            false, false);
    }

#pragma unroll
    for (int j = 0; j < NB; ++j) {
        int n = (ng * NB + j) * 16 + lr;
        float bn = (flags & GF_BIAS) ? bias[n] : 0.0f;
#pragma unroll
        for (int r = 0; r < 8; ++r) {
            int m = mt * 16 + r + 8 * hi;
            float v = acc[j][r] * alpha + bn;
            if (flags & GF_RES)  v += res[(size_t)m * ldres + n];
            if (flags & GF_GELU) v = 0.5f * v * (1.0f + erff(v * 0.70710678f));
            if (flags & GF_RELU) v = v > 0.0f ? v : 0.0f;
            if (n < Nvalid) {
                if (C) {
                    if (flags & GF_TRANSC) C[(size_t)n * ldc + m] = v;
                    else                   C[(size_t)m * ldc + n] = v;
                }
                if (Ch) Ch[(size_t)m * ldch + n] = (_Float16)v;
            }
        }
    }
}

// ---------------------------------------------------------------------------
// Implicit-GEMM dilated 3x3x3 conv, C=128 -> 128, scene 64x64x8, residual+ReLU.
// X layout: [voxel][128]; Wt layout: [tap][O][I] (pre-transposed).  One wave:
// 16 voxels x 64 output channels (4 accumulators), 27 taps x 4 K-steps x 4 =
// 432 WMMAs.  Out-of-bounds neighbors read a zeroed guard row (branch-free,
// no EXEC divergence, loads stay pipelined).
// ---------------------------------------------------------------------------
__global__ __launch_bounds__(256)
void k_conv3d_wmma(const _Float16* __restrict__ Xh,
                   const float* __restrict__ Xf,
                   const _Float16* __restrict__ Wt,
                   const float* __restrict__ bias,
                   const _Float16* __restrict__ zrow,
                   float* __restrict__ Yf,
                   _Float16* __restrict__ Yh,
                   int dil)
{
    int lane = threadIdx.x & 31;
    int wid  = threadIdx.x >> 5;
    int tile = blockIdx.x * (blockDim.x >> 5) + wid;   // (2048 mt) * (2 ngroups)
    if (tile >= (NQTOT / 16) * 2) return;
    int mt = tile >> 1, ng = tile & 1;
    int lr = lane & 15, hi = lane >> 4;

    int vox = mt * 16 + lr;
    int z = vox & 7, y = (vox >> 3) & 63, x = (vox >> 9) & 63;

    v8f acc[4];
#pragma unroll
    for (int j = 0; j < 4; ++j) acc[j] = (v8f){};

    for (int tap = 0; tap < 27; ++tap) {
        int kw = tap % 3, kh = (tap / 3) % 3, kd = tap / 9;
        int nx = x + (kd - 1) * dil;
        int ny = y + (kh - 1) * dil;
        int nz = z + (kw - 1) * dil;
        bool ok = ((unsigned)nx < 64u) && ((unsigned)ny < 64u) && ((unsigned)nz < 8u);
        int nm = ((nx * 64) + ny) * 8 + nz;
        const _Float16* arow = (ok ? Xh + (size_t)nm * CDIM : zrow) + 16 * hi;
        const _Float16* brow[4];
#pragma unroll
        for (int j = 0; j < 4; ++j)
            brow[j] = Wt + (size_t)tap * (CDIM * CDIM)
                         + (size_t)((ng * 4 + j) * 16 + lr) * CDIM + 16 * hi;
#pragma unroll
        for (int kb = 0; kb < CDIM; kb += 32) {
            v8h a0 = *(const v8h*)(arow + kb);
            v8h a1 = *(const v8h*)(arow + kb + 8);
            v16h bv[4];
#pragma unroll
            for (int j = 0; j < 4; ++j) bv[j] = *(const v16h*)(brow[j] + kb);
            v16h av = {};
#pragma unroll
            for (int i = 0; i < 8; ++i) { av[i] = a0[i]; av[8 + i] = a1[i]; }
#pragma unroll
            for (int j = 0; j < 4; ++j)
                acc[j] = __builtin_amdgcn_wmma_f32_16x16x32_f16(false, av, false, bv[j],
                                                                (short)0, acc[j],
                                                                false, false);
        }
    }

#pragma unroll
    for (int j = 0; j < 4; ++j) {
        int n = (ng * 4 + j) * 16 + lr;
#pragma unroll
        for (int r = 0; r < 8; ++r) {
            int m = mt * 16 + r + 8 * hi;
            float v = acc[j][r] + bias[n] + Xf[(size_t)m * CDIM + n];
            v = v > 0.0f ? v : 0.0f;
            Yf[(size_t)m * CDIM + n] = v;
            Yh[(size_t)m * CDIM + n] = (_Float16)v;
        }
    }
}

// ------------------------- small helper kernels ----------------------------
__global__ void k_f32_to_f16(const float* __restrict__ in, _Float16* __restrict__ out, int n)
{
    int i = blockIdx.x * blockDim.x + threadIdx.x;
    if (i < n) out[i] = (_Float16)in[i];
}

__global__ void k_copy_f32(const float* __restrict__ in, float* __restrict__ out, int n)
{
    int i = blockIdx.x * blockDim.x + threadIdx.x;
    if (i < n) out[i] = in[i];
}

__global__ void k_zero_i32(int* p, int n)
{
    int i = blockIdx.x * blockDim.x + threadIdx.x;
    if (i < n) p[i] = 0;
}

// per-instance class softmax of sigmoid(logits)/T, score=max, keep0
__global__ void k_inst_scores(const float* __restrict__ logits,
                              float* __restrict__ score, int* __restrict__ keep0)
{
    int t = threadIdx.x;
    if (t < NREAL) {
        float v[NCLS]; float mx = -1e30f;
        for (int c = 0; c < NCLS; ++c) {
            float l = logits[t * NCLS + c];
            float s = 1.0f / (1.0f + expf(-l));
            v[c] = s / 0.06f;
            mx = fmaxf(mx, v[c]);
        }
        float sum = 0.0f;
        for (int c = 0; c < NCLS; ++c) sum += expf(v[c] - mx);
        float best = 0.0f;
        for (int c = 0; c < NCLS; ++c) best = fmaxf(best, expf(v[c] - mx) / sum);
        score[t] = best;
        keep0[t] = best > 0.25f ? 1 : 0;
    } else if (t < NP) { score[t] = 0.0f; keep0[t] = 0; }
}

__global__ void k_mask_ids(const float* __restrict__ pm,
                           const float* __restrict__ score,
                           const int* __restrict__ keep0,
                           int* __restrict__ ids)
{
    int p = blockIdx.x * blockDim.x + threadIdx.x;
    if (p >= HWPIX) return;
    float best = -1e30f; int bi = 0;
    for (int n = 0; n < NREAL; ++n) {
        float s = 1.0f / (1.0f + expf(-pm[(size_t)n * HWPIX + p]));
        float q = keep0[n] ? score[n] * s : -1.0f;
        if (q > best) { best = q; bi = n; }   // first-max (jnp.argmax) tie rule
    }
    ids[p] = bi;
}

__global__ void k_areas(const float* __restrict__ pm,
                        const int* __restrict__ keep0,
                        const int* __restrict__ ids,
                        int* __restrict__ pm_area, int* __restrict__ mask_area,
                        int* __restrict__ inter, unsigned char* __restrict__ maskbit)
{
    int idx = blockIdx.x * blockDim.x + threadIdx.x;
    if (idx >= NREAL * HWPIX) return;
    int n = idx / HWPIX, p = idx % HWPIX;
    float s = 1.0f / (1.0f + expf(-pm[idx]));
    int pmb = s >= 0.5f;
    int m0 = (ids[p] == n) && keep0[n];
    if (pmb) atomicAdd(&pm_area[n], 1);
    if (m0)  atomicAdd(&mask_area[n], 1);
    int mb = m0 && pmb;
    if (mb)  atomicAdd(&inter[n], 1);
    maskbit[idx] = (unsigned char)mb;
}

__global__ void k_finalize(const int* __restrict__ keep0, const int* __restrict__ inter,
                           const int* __restrict__ mask_area, const int* __restrict__ pm_area,
                           int* __restrict__ keep, float* __restrict__ attnbias,
                           int* __restrict__ anykeep)
{
    int t = threadIdx.x;
    if (t < NREAL) {
        keep[t] = (keep0[t] && inter[t] > 0 &&
                   (float)mask_area[t] >= 0.8f * (float)pm_area[t]) ? 1 : 0;
    } else if (t < NP) keep[t] = 0;
    __syncthreads();
    if (t == 0) {
        int any = 0;
        for (int n = 0; n < NREAL; ++n) any |= keep[n];
        anykeep[0] = any;
    }
    __syncthreads();
    int any = anykeep[0];
    if (t < NREAL) attnbias[t] = (!any || keep[t]) ? 0.0f : -1e9f;
    else if (t < NP) attnbias[t] = -1e9f;
}

__global__ void k_inverses(const float* __restrict__ Kin, const float* __restrict__ Ein,
                           float* __restrict__ invK, float* __restrict__ invE)
{
    // 3x3 adjugate inverse
    float a = Kin[0], b = Kin[1], c = Kin[2];
    float d = Kin[3], e = Kin[4], f = Kin[5];
    float g = Kin[6], h = Kin[7], i = Kin[8];
    float det = a * (e * i - f * h) - b * (d * i - f * g) + c * (d * h - e * g);
    float id = 1.0f / det;
    invK[0] = (e * i - f * h) * id; invK[1] = (c * h - b * i) * id; invK[2] = (b * f - c * e) * id;
    invK[3] = (f * g - d * i) * id; invK[4] = (a * i - c * g) * id; invK[5] = (c * d - a * f) * id;
    invK[6] = (d * h - e * g) * id; invK[7] = (b * g - a * h) * id; invK[8] = (a * e - b * d) * id;
    // 4x4 Gauss-Jordan with partial pivoting
    float m[4][8];
    for (int r = 0; r < 4; ++r)
        for (int cc = 0; cc < 4; ++cc) { m[r][cc] = Ein[r * 4 + cc]; m[r][4 + cc] = (r == cc) ? 1.0f : 0.0f; }
    for (int cc = 0; cc < 4; ++cc) {
        int pr = cc;
        for (int r = cc + 1; r < 4; ++r) if (fabsf(m[r][cc]) > fabsf(m[pr][cc])) pr = r;
        if (pr != cc) for (int j = 0; j < 8; ++j) { float t = m[cc][j]; m[cc][j] = m[pr][j]; m[pr][j] = t; }
        float piv = 1.0f / m[cc][cc];
        for (int j = 0; j < 8; ++j) m[cc][j] *= piv;
        for (int r = 0; r < 4; ++r) if (r != cc) {
            float fct = m[r][cc];
            for (int j = 0; j < 8; ++j) m[r][j] -= fct * m[cc][j];
        }
    }
    for (int r = 0; r < 4; ++r)
        for (int cc = 0; cc < 4; ++cc) invE[r * 4 + cc] = m[r][4 + cc];
}

__global__ void k_pts(const float* __restrict__ depth, const float* __restrict__ invK,
                      const float* __restrict__ invE, float* __restrict__ pts)
{
    int p = blockIdx.x * blockDim.x + threadIdx.x;
    if (p >= HWPIX) return;
    int r = p / IMW, c = p % IMW;
    float xi = c * (1279.0f / 159.0f);
    float yi = r * (383.0f / 47.0f);
    float d = depth[p];
    float g0 = xi * d, g1 = yi * d, g2 = d;
    float cx = invK[0] * g0 + invK[1] * g1 + invK[2] * g2;
    float cy = invK[3] * g0 + invK[4] * g1 + invK[5] * g2;
    float cz = invK[6] * g0 + invK[7] * g1 + invK[8] * g2;
    for (int j = 0; j < 3; ++j)
        pts[p * 3 + j] = invE[j * 4 + 0] * cx + invE[j * 4 + 1] * cy +
                         invE[j * 4 + 2] * cz + invE[j * 4 + 3];
}

__global__ void k_inst_xyz(const unsigned char* __restrict__ maskbit,
                           const int* __restrict__ keep,
                           const float* __restrict__ pts, float* __restrict__ xyz)
{
    __shared__ float sx[256], sy[256], sz[256];
    int n = blockIdx.x, t = threadIdx.x;
    float ax = 0, ay = 0, az = 0;
    if (keep[n]) {
        for (int p = t; p < HWPIX; p += 256)
            if (maskbit[(size_t)n * HWPIX + p]) {
                ax += pts[p * 3]; ay += pts[p * 3 + 1]; az += pts[p * 3 + 2];
            }
    }
    sx[t] = ax; sy[t] = ay; sz[t] = az;
    __syncthreads();
    for (int o = 128; o > 0; o >>= 1) {
        if (t < o) { sx[t] += sx[t + o]; sy[t] += sy[t + o]; sz[t] += sz[t + o]; }
        __syncthreads();
    }
    if (t == 0) { xyz[n * 3] = sx[0]; xyz[n * 3 + 1] = sy[0]; xyz[n * 3 + 2] = sz[0]; }
}

__global__ void k_instpos(const float* __restrict__ xyz, const float* __restrict__ pw,
                          float* __restrict__ instpos)
{
    int idx = blockIdx.x * blockDim.x + threadIdx.x;
    if (idx >= NP * CDIM) return;
    int n = idx / CDIM, c = idx % CDIM;
    float v = 0.0f;
    if (n < NREAL)
        for (int j = 0; j < 3; ++j) v += xyz[n * 3 + j] * pw[c * 3 + j];
    instpos[idx] = v;
}

__global__ void k_kvprep(const float* __restrict__ iq, const float* __restrict__ instpos,
                         const int* __restrict__ anykeep,
                         _Float16* __restrict__ kvk16, _Float16* __restrict__ iq16)
{
    int idx = blockIdx.x * blockDim.x + threadIdx.x;
    if (idx >= NP * CDIM) return;
    int n = idx / CDIM;
    if (n < NREAL) {
        float u = anykeep[0] ? 1.0f : 0.0f;
        float q = iq[idx];
        kvk16[idx] = (_Float16)(q + u * instpos[idx]);
        iq16[idx]  = (_Float16)q;
    } else { kvk16[idx] = (_Float16)0.0f; iq16[idx] = (_Float16)0.0f; }
}

__global__ void k_qpos(const float* __restrict__ vox, const float* __restrict__ pw,
                       float* __restrict__ qpos)
{
    int idx = blockIdx.x * blockDim.x + threadIdx.x;
    if (idx >= NQTOT * CDIM) return;
    int q = idx / CDIM, c = idx % CDIM;
    int z = q & 7, y = (q >> 3) & 63, x = (q >> 9) & 63;
    float p0 = (x + 0.5f) * 0.8f + vox[0];
    float p1 = (y + 0.5f) * 0.8f + vox[1];
    float p2 = (z + 0.5f) * 0.8f + vox[2];
    qpos[idx] = p0 * pw[c * 3 + 0] + p1 * pw[c * 3 + 1] + p2 * pw[c * 3 + 2];
}

__global__ void k_addpos(const float* __restrict__ q, const float* __restrict__ qpos,
                         float* __restrict__ x, _Float16* __restrict__ xh, int n)
{
    int i = blockIdx.x * blockDim.x + threadIdx.x;
    if (i >= n) return;
    float v = q[i] + qpos[i];
    x[i] = v; xh[i] = (_Float16)v;
}

__global__ void k_softmax_row(const float* __restrict__ s, _Float16* __restrict__ p)
{
    __shared__ float red[CDIM];
    size_t row = blockIdx.x; int t = threadIdx.x;
    float v = s[row * CDIM + t];
    red[t] = v; __syncthreads();
    for (int o = 64; o > 0; o >>= 1) { if (t < o) red[t] = fmaxf(red[t], red[t + o]); __syncthreads(); }
    float mx = red[0]; __syncthreads();
    float e = expf(v - mx);
    red[t] = e; __syncthreads();
    for (int o = 64; o > 0; o >>= 1) { if (t < o) red[t] += red[t + o]; __syncthreads(); }
    p[row * CDIM + t] = (_Float16)(e / red[0]);
}

__global__ void k_layernorm(const float* __restrict__ in, const float* __restrict__ g,
                            const float* __restrict__ b, float* __restrict__ out,
                            _Float16* __restrict__ outh)
{
    __shared__ float red[CDIM];
    size_t row = blockIdx.x; int t = threadIdx.x;
    float v = in[row * CDIM + t];
    red[t] = v; __syncthreads();
    for (int o = 64; o > 0; o >>= 1) { if (t < o) red[t] += red[t + o]; __syncthreads(); }
    float mean = red[0] * (1.0f / CDIM); __syncthreads();
    float d = v - mean;
    red[t] = d * d; __syncthreads();
    for (int o = 64; o > 0; o >>= 1) { if (t < o) red[t] += red[t + o]; __syncthreads(); }
    float var = red[0] * (1.0f / CDIM);
    float y = d * rsqrtf(var + 1e-5f) * g[t] + b[t];
    out[row * CDIM + t] = y;
    if (outh) outh[row * CDIM + t] = (_Float16)y;
}

__global__ void k_transpose128(const _Float16* __restrict__ in, _Float16* __restrict__ out)
{
    int idx = blockIdx.x * blockDim.x + threadIdx.x;
    if (idx >= CDIM * NP) return;
    int c = idx / NP, n = idx % NP;
    out[c * NP + n] = in[n * CDIM + c];
}

__global__ void k_convw(const float* __restrict__ w, _Float16* __restrict__ wt)
{
    int idx = blockIdx.x * blockDim.x + threadIdx.x;
    if (idx >= CDIM * CDIM * 27) return;
    int tap = idx % 27; int rem = idx / 27;
    int i = rem % CDIM, o = rem / CDIM;
    wt[(size_t)tap * (CDIM * CDIM) + o * CDIM + i] = (_Float16)w[idx];
}

__global__ void k_outw(const float* __restrict__ w, const float* __restrict__ b,
                       _Float16* __restrict__ wp, float* __restrict__ bp)
{
    int idx = blockIdx.x * blockDim.x + threadIdx.x;
    if (idx >= 32 * CDIM) return;
    int o = idx / CDIM, i = idx % CDIM;
    wp[idx] = (o < 20) ? (_Float16)w[o * CDIM + i] : (_Float16)0.0f;
    if (i == 0) bp[o] = (o < 20) ? b[o] : 0.0f;
}

__global__ void k_fov(const float* __restrict__ q, const float* __restrict__ qe,
                      const unsigned char* __restrict__ fov,
                      float* __restrict__ xf, _Float16* __restrict__ xh, int n)
{
    int idx = blockIdx.x * blockDim.x + threadIdx.x;
    if (idx >= n) return;
    int v = idx / CDIM;
    float val = fov[v] ? q[idx] : qe[idx];
    xf[idx] = val; xh[idx] = (_Float16)val;
}

// ---------------------------------------------------------------------------
extern "C" void kernel_launch(void* const* d_in, const int* in_sizes, int n_in,
                              void* d_out, int out_size, void* d_ws, size_t ws_size,
                              hipStream_t stream)
{
    const float* inst_queries = (const float*)d_in[0];
    const float* pred_logits  = (const float*)d_in[1];
    const float* pred_masks   = (const float*)d_in[2];
    const float* depth        = (const float*)d_in[3];
    const float* Kin          = (const float*)d_in[4];
    const float* Ein          = (const float*)d_in[5];
    const float* vox_origin   = (const float*)d_in[6];
    const float* query_embed  = (const float*)d_in[7];
    const float* pos_embed    = (const float*)d_in[8];
    const float* attn_in_w    = (const float*)d_in[9];
    const float* attn_in_b    = (const float*)d_in[10];
    const float* attn_out_w   = (const float*)d_in[11];
    const float* attn_out_b   = (const float*)d_in[12];
    const float* ln1_g        = (const float*)d_in[13];
    const float* ln1_b        = (const float*)d_in[14];
    const float* ffn_w1       = (const float*)d_in[15];
    const float* ffn_b1       = (const float*)d_in[16];
    const float* ffn_w2       = (const float*)d_in[17];
    const float* ffn_b2       = (const float*)d_in[18];
    const float* ln2_g        = (const float*)d_in[19];
    const float* ln2_b        = (const float*)d_in[20];
    const float* proc_w       = (const float*)d_in[21];
    const float* proc_b       = (const float*)d_in[22];
    const float* out_w        = (const float*)d_in[23];
    const float* out_b        = (const float*)d_in[24];
    const unsigned char* fov  = (const unsigned char*)d_in[25];
    float* dout = (float*)d_out;

    // ---- workspace layout (256B aligned, aggressive aliasing) ----
    size_t off = 0;
    auto alloc = [&](size_t bytes) -> char* {
        char* p = (char*)d_ws + off;
        off += (bytes + 255) & ~(size_t)255;
        return p;
    };
    const size_t B32 = (size_t)NQTOT * CDIM * 4;   // 16 MB
    const size_t B16 = (size_t)NQTOT * CDIM * 2;   //  8 MB

    float*    qpos   = (float*)alloc(B32);
    float*    qbuf   = (float*)alloc(B32);
    float*    xbuf   = (float*)alloc(B32);
    float*    attn   = (float*)alloc(B32);          // also FFN2 output "y"
    float*    x2     = (float*)alloc(B32);
    float*    scores = (float*)alloc(4 * B32);      // [4][NQ][128] f32
    _Float16* ffn16  = (_Float16*)scores;           // alias (32MB of 64MB)
    float*    xbF    = (float*)((char*)scores + 2 * B32);  // conv ping-pong f32 (alias)
    _Float16* probs  = (_Float16*)alloc(4 * B16);   // [4][NQ][128] f16
    float*    xaF    = (float*)probs;               // conv f32 (alias, 16MB)
    _Float16* xaH    = (_Float16*)((char*)probs + 2 * B16); // conv f16 (alias, 8MB)
    _Float16* x16    = (_Float16*)alloc(B16);
    _Float16* Q16    = (_Float16*)alloc(B16);
    _Float16* xbH    = Q16;                         // conv ping-pong f16 (alias)
    _Float16* O16    = (_Float16*)alloc(B16);
    _Float16* x216   = (_Float16*)alloc(B16);

    _Float16* wIn16  = (_Float16*)alloc(384 * CDIM * 2);
    _Float16* wo16   = (_Float16*)alloc(CDIM * CDIM * 2);
    _Float16* w116   = (_Float16*)alloc(FFND * CDIM * 2);
    _Float16* w216   = (_Float16*)alloc(CDIM * FFND * 2);
    _Float16* wtap   = (_Float16*)alloc(27 * CDIM * CDIM * 2);
    _Float16* wout16 = (_Float16*)alloc(32 * CDIM * 2);
    float*    boutP  = (float*)alloc(32 * 4);
    _Float16* zrow   = (_Float16*)alloc(CDIM * 2);  // zero guard row (conv OOB)

    _Float16* kvk16  = (_Float16*)alloc(NP * CDIM * 2);
    _Float16* iq16   = (_Float16*)alloc(NP * CDIM * 2);
    _Float16* K16    = (_Float16*)alloc(NP * CDIM * 2);
    _Float16* V16    = (_Float16*)alloc(NP * CDIM * 2);
    _Float16* Vt16   = (_Float16*)alloc(NP * CDIM * 2);

    float*    scoreI = (float*)alloc(NP * 4);
    int*      keep0  = (int*)alloc(NP * 4);
    int*      keep   = (int*)alloc(NP * 4);
    int*      cnt    = (int*)alloc(3 * NP * 4);     // pm_area / mask_area / inter
    int*      pmA = cnt, *mkA = cnt + NP, *itA = cnt + 2 * NP;
    unsigned char* maskbit = (unsigned char*)alloc((size_t)NREAL * HWPIX);
    int*      ids    = (int*)alloc(HWPIX * 4);
    float*    pts    = (float*)alloc(HWPIX * 3 * 4);
    float*    invK   = (float*)alloc(16 * 4);
    float*    invE   = (float*)alloc(16 * 4);
    float*    xyz    = (float*)alloc(NREAL * 3 * 4);
    float*    instpos= (float*)alloc(NP * CDIM * 4);
    float*    abias  = (float*)alloc(NP * 4);
    int*      anyk   = (int*)alloc(64 * 4);
    (void)ws_size; (void)n_in; (void)in_sizes; (void)out_size;

    // NB = N-register-blocking factor (tilesN must be divisible by NB)
    auto gemm = [&](const _Float16* A, int lda, const _Float16* B, int ldb,
                    const float* bias, const float* res, int ldres,
                    float* C, int ldc, _Float16* Ch, int ldch,
                    int M, int K, int tilesN, int NB, int Nvalid,
                    float alpha, int flags) {
        int ng = tilesN / NB;
        int tiles = (M >> 4) * ng;
        int blocks = (tiles + 7) / 8;
        switch (NB) {
        case 1:
            k_gemm_wmma<1><<<blocks, 256, 0, stream>>>(A, lda, B, ldb, bias, res, ldres,
                                                       C, ldc, Ch, ldch, M, K, ng,
                                                       Nvalid, alpha, flags);
            break;
        case 2:
            k_gemm_wmma<2><<<blocks, 256, 0, stream>>>(A, lda, B, ldb, bias, res, ldres,
                                                       C, ldc, Ch, ldch, M, K, ng,
                                                       Nvalid, alpha, flags);
            break;
        default:
            k_gemm_wmma<4><<<blocks, 256, 0, stream>>>(A, lda, B, ldb, bias, res, ldres,
                                                       C, ldc, Ch, ldch, M, K, ng,
                                                       Nvalid, alpha, flags);
            break;
        }
    };
    auto cvt = [&](const float* src, _Float16* dst, int n) {
        k_f32_to_f16<<<(n + 255) / 256, 256, 0, stream>>>(src, dst, n);
    };

    // ---- panoptic post-processing + back-projection ----
    k_inverses<<<1, 1, 0, stream>>>(Kin, Ein, invK, invE);
    k_pts<<<(HWPIX + 255) / 256, 256, 0, stream>>>(depth, invK, invE, pts);
    k_inst_scores<<<1, 128, 0, stream>>>(pred_logits, scoreI, keep0);
    k_mask_ids<<<(HWPIX + 255) / 256, 256, 0, stream>>>(pred_masks, scoreI, keep0, ids);
    k_zero_i32<<<2, 256, 0, stream>>>(cnt, 3 * NP);
    k_zero_i32<<<1, 64, 0, stream>>>((int*)zrow, CDIM / 2);
    k_areas<<<(NREAL * HWPIX + 255) / 256, 256, 0, stream>>>(pred_masks, keep0, ids,
                                                             pmA, mkA, itA, maskbit);
    k_finalize<<<1, 128, 0, stream>>>(keep0, itA, mkA, pmA, keep, abias, anyk);
    k_inst_xyz<<<NREAL, 256, 0, stream>>>(maskbit, keep, pts, xyz);
    k_instpos<<<(NP * CDIM + 255) / 256, 256, 0, stream>>>(xyz, pos_embed, instpos);
    k_kvprep<<<(NP * CDIM + 255) / 256, 256, 0, stream>>>(inst_queries, instpos, anyk,
                                                          kvk16, iq16);

    // ---- positional embedding of scene queries, q init ----
    k_qpos<<<(NQTOT * CDIM + 255) / 256, 256, 0, stream>>>(vox_origin, pos_embed, qpos);
    k_copy_f32<<<(NQTOT * CDIM + 255) / 256, 256, 0, stream>>>(query_embed, qbuf,
                                                               NQTOT * CDIM);

    // ---- 3 transformer decoder layers (cross-attn over 128-padded keys) ----
    const float scale = 0.1767766952966369f;   // 1/sqrt(32)
    for (int li = 0; li < 3; ++li) {
        cvt(attn_in_w + (size_t)li * 384 * CDIM, wIn16, 384 * CDIM);
        cvt(attn_out_w + (size_t)li * CDIM * CDIM, wo16, CDIM * CDIM);
        cvt(ffn_w1 + (size_t)li * FFND * CDIM, w116, FFND * CDIM);
        cvt(ffn_w2 + (size_t)li * CDIM * FFND, w216, CDIM * FFND);
        const float* bIn = attn_in_b + (size_t)li * 384;

        k_addpos<<<(NQTOT * CDIM + 255) / 256, 256, 0, stream>>>(qbuf, qpos, xbuf, x16,
                                                                 NQTOT * CDIM);
        // Q/K/V projections
        gemm(x16, CDIM, wIn16, CDIM, bIn, nullptr, 0, nullptr, 0, Q16, CDIM,
             NQTOT, CDIM, 8, 4, CDIM, 1.0f, GF_BIAS);
        gemm(kvk16, CDIM, wIn16 + CDIM * CDIM, CDIM, bIn + CDIM, nullptr, 0,
             nullptr, 0, K16, CDIM, NP, CDIM, 8, 4, CDIM, 1.0f, GF_BIAS);
        gemm(iq16, CDIM, wIn16 + 2 * CDIM * CDIM, CDIM, bIn + 2 * CDIM, nullptr, 0,
             nullptr, 0, V16, CDIM, NP, CDIM, 8, 4, CDIM, 1.0f, GF_BIAS);
        k_transpose128<<<(NP * CDIM + 255) / 256, 256, 0, stream>>>(V16, Vt16);
        // attention scores per head (+mask bias), softmax, apply
        for (int h = 0; h < 4; ++h)
            gemm(Q16 + h * 32, CDIM, K16 + h * 32, CDIM, abias, nullptr, 0,
                 scores + (size_t)h * NQTOT * CDIM, CDIM, nullptr, 0,
                 NQTOT, 32, 8, 4, CDIM, scale, GF_BIAS);
        k_softmax_row<<<4 * NQTOT, 128, 0, stream>>>(scores, probs);
        for (int h = 0; h < 4; ++h)
            gemm(probs + (size_t)h * NQTOT * CDIM, CDIM, Vt16 + h * 32 * NP, NP,
                 nullptr, nullptr, 0, nullptr, 0, O16 + h * 32, CDIM,
                 NQTOT, NP, 2, 2, 32, 1.0f, 0);
        // output projection + residual, LN1
        gemm(O16, CDIM, wo16, CDIM, attn_out_b + (size_t)li * CDIM, xbuf, CDIM,
             attn, CDIM, nullptr, 0, NQTOT, CDIM, 8, 4, CDIM, 1.0f, GF_BIAS | GF_RES);
        k_layernorm<<<NQTOT, 128, 0, stream>>>(attn, ln1_g + (size_t)li * CDIM,
                                               ln1_b + (size_t)li * CDIM, x2, x216);
        // FFN (GELU) + residual, LN2 -> q
        gemm(x216, CDIM, w116, CDIM, ffn_b1 + (size_t)li * FFND, nullptr, 0,
             nullptr, 0, ffn16, FFND, NQTOT, CDIM, 32, 4, FFND, 1.0f, GF_BIAS | GF_GELU);
        gemm(ffn16, FFND, w216, FFND, ffn_b2 + (size_t)li * CDIM, x2, CDIM,
             attn, CDIM, nullptr, 0, NQTOT, FFND, 8, 4, CDIM, 1.0f, GF_BIAS | GF_RES);
        k_layernorm<<<NQTOT, 128, 0, stream>>>(attn, ln2_g + (size_t)li * CDIM,
                                               ln2_b + (size_t)li * CDIM, qbuf, nullptr);
    }

    // ---- fov scatter: q where fov_mask else query_embed ----
    k_fov<<<(NQTOT * CDIM + 255) / 256, 256, 0, stream>>>(qbuf, query_embed, fov,
                                                          xaF, xaH, NQTOT * CDIM);

    // ---- 3 dilated residual 3D convs (implicit GEMM, WMMA) ----
    float* curF = xaF; _Float16* curH = xaH;
    float* nxtF = xbF; _Float16* nxtH = xbH;
    const int dils[3] = {1, 2, 3};
    for (int i = 0; i < 3; ++i) {
        k_convw<<<(27 * CDIM * CDIM + 255) / 256, 256, 0, stream>>>(
            proc_w + (size_t)i * 27 * CDIM * CDIM, wtap);
        k_conv3d_wmma<<<512, 256, 0, stream>>>(curH, curF, wtap,
                                               proc_b + (size_t)i * CDIM, zrow,
                                               nxtF, nxtH, dils[i]);
        float* tf = curF; curF = nxtF; nxtF = tf;
        _Float16* th = curH; curH = nxtH; nxtH = th;
    }

    // ---- final 1x1x1 conv -> d_out ([20][32768] channel-major) ----
    k_outw<<<(32 * CDIM + 255) / 256, 256, 0, stream>>>(out_w, out_b, wout16, boutP);
    gemm(curH, CDIM, wout16, CDIM, boutP, nullptr, 0, dout, NQTOT, nullptr, 0,
         NQTOT, CDIM, 2, 2, 20, 1.0f, GF_BIAS | GF_TRANSC);
}
